// GAT_75213467287865
// MI455X (gfx1250) — matrix operand order repaired and checked by reading, not compiled, not run on hardware
//
#include <hip/hip_runtime.h>
#include <hip/hip_bf16.h>

#define NN 50000
#define NE 800000

typedef __attribute__((ext_vector_type(16))) _Float16 v16h;
typedef __attribute__((ext_vector_type(8)))  _Float16 v8h;
typedef __attribute__((ext_vector_type(8)))  float    v8f;

// ---------------------------------------------------------------------------
// GEMM: C[N,M] (f32) = A[N,K] (f16, row-major) x Bt[M,K]^T (f16, row-major)
// One wave computes a 16-row strip of MT 16x16 tiles via v_wmma_f32_16x16x32_f16.
// Fragment layouts used here (wave32; A/B use contiguous k order, not the ISA table):
//   A: lane l holds row (l&15); half j -> K=(l>>4)*16+j   (contiguous in A row)
//   B: lane l holds col (l&15); half j -> K=(l>>4)*16+j   (contiguous in Bt row)
//   C: lane l holds col (l&15); vgpr i -> row (l>>4)*8+i
// ---------------------------------------------------------------------------
template<int MT>
__global__ __launch_bounds__(256)
void k_gemm_wmma(const _Float16* __restrict__ A,
                 const _Float16* __restrict__ Bt,
                 float* __restrict__ C,
                 int N, int K, int M) {
  const int gwave = (int)((blockIdx.x * blockDim.x + threadIdx.x) >> 5);
  const int lane  = (int)(threadIdx.x & 31);
  const int colGroups = M / (16 * MT);
  const int rowTile   = gwave / colGroups;
  const int colGroup  = gwave - rowTile * colGroups;
  if (rowTile * 16 >= N) return;           // wave-uniform: EXEC stays all-ones for WMMA
  const int row0 = rowTile * 16;
  const int lm = lane & 15;
  const int lk = lane >> 4;

  v8f acc[MT] = {};

  const _Float16* aBase = A + (size_t)(row0 + lm) * K + lk * 16;
  for (int k0 = 0; k0 < K; k0 += 32) {
    union { v16h v; v8h h[2]; } au;
    au.h[0] = *(const v8h*)(aBase + k0);        // K = k0 + lk*16 + 0..7
    au.h[1] = *(const v8h*)(aBase + k0 + 8);    // K = k0 + lk*16 + 8..15
#pragma unroll
    for (int t = 0; t < MT; ++t) {
      const int col0 = (colGroup * MT + t) * 16;
      const _Float16* bp = Bt + (size_t)(col0 + lm) * K + k0 + lk * 16;
      v16h b = *(const v16h*)bp;                // K = k0 + lk*16 + 0..15 (32B contiguous)
      acc[t] = __builtin_amdgcn_wmma_f32_16x16x32_f16(
          false, au.v, false, b, (short)0, acc[t], false, false);
    }
  }
#pragma unroll
  for (int t = 0; t < MT; ++t) {
    const int col0 = (colGroup * MT + t) * 16;
#pragma unroll
    for (int i = 0; i < 8; ++i)
      C[(size_t)(row0 + lk * 8 + i) * M + col0 + lm] = acc[t][i];
  }
}

// ---------------------------------------------------------------------------
// Helpers / element-wise kernels
// ---------------------------------------------------------------------------
__global__ void k_f32_to_f16(const float* __restrict__ x, _Float16* __restrict__ y, int n) {
  int i = blockIdx.x * blockDim.x + threadIdx.x;
  if (i < n) y[i] = (_Float16)x[i];
}

__global__ void k_transpose_f16(const float* __restrict__ W, _Float16* __restrict__ Wt,
                                int K, int M) {
  int i = blockIdx.x * blockDim.x + threadIdx.x;
  if (i < K * M) { int k = i / M, m = i - k * M; Wt[(size_t)m * K + k] = (_Float16)W[i]; }
}

// el[n,h] = sum_d feat[n,h,d]*al[h,d]; er likewise
__global__ void k_elr(const float* __restrict__ feat, const float* __restrict__ al,
                      const float* __restrict__ ar, float* __restrict__ el,
                      float* __restrict__ er, int N, int H, int D) {
  int i = blockIdx.x * blockDim.x + threadIdx.x;
  if (i >= N * H) return;
  int n = i / H, h = i - n * H;
  const float* f = feat + (size_t)n * H * D + h * D;
  const float* a = al + h * D;
  const float* r = ar + h * D;
  float sl = 0.f, sr = 0.f;
  for (int d = 0; d < D; ++d) { float x = f[d]; sl += x * a[d]; sr += x * r[d]; }
  el[i] = sl; er[i] = sr;
}

__global__ void k_init_ms(float* __restrict__ mm, float* __restrict__ ss, int n) {
  int i = blockIdx.x * blockDim.x + threadIdx.x;
  if (i < n) { mm[i] = -__builtin_inff(); ss[i] = 0.f; }
}

// out[n*M + j] = bias[j]  (M is a power of two -> mask)
__global__ void k_init_out(const float* __restrict__ b, float* __restrict__ o,
                           int mmask, int total) {
  int i = blockIdx.x * blockDim.x + threadIdx.x;
  if (i < total) o[i] = b[i & mmask];
}

// sign-aware float atomic max via integer atomics (init must be -inf)
__device__ inline void atomicMaxF(float* addr, float v) {
  if (v >= 0.f) atomicMax((int*)addr, __float_as_int(v));
  else          atomicMin((unsigned int*)addr, __float_as_uint(v));
}

__global__ void k_edge_max(const int* __restrict__ src, const int* __restrict__ dst,
                           const float* __restrict__ el, const float* __restrict__ er,
                           float* __restrict__ mm, int total, int hs, int hmask) {
  int i = blockIdx.x * blockDim.x + threadIdx.x;
  if (i >= total) return;
  int e = i >> hs, h = i & hmask;
  int s = src[e], d = dst[e];
  float v = el[(s << hs) + h] + er[(d << hs) + h];
  v = v > 0.f ? v : 0.2f * v;                 // leaky_relu, slope 0.2
  atomicMaxF(&mm[(d << hs) + h], v);
}

__global__ void k_edge_exp(const int* __restrict__ src, const int* __restrict__ dst,
                           const float* __restrict__ el, const float* __restrict__ er,
                           const float* __restrict__ mm, float* __restrict__ ss,
                           float* __restrict__ ee, int total, int hs, int hmask) {
  int i = blockIdx.x * blockDim.x + threadIdx.x;
  if (i >= total) return;
  int e = i >> hs, h = i & hmask;
  int s = src[e], d = dst[e];
  float v = el[(s << hs) + h] + er[(d << hs) + h];
  v = v > 0.f ? v : 0.2f * v;
  float x = expf(v - mm[(d << hs) + h]);
  ee[i] = x;
  atomicAdd(&ss[(d << hs) + h], x);
}

// one thread per (edge, 4 features): out[dst] += feat[src] * alpha
__global__ void k_edge_agg(const int* __restrict__ src, const int* __restrict__ dst,
                           const float* __restrict__ feat, const float* __restrict__ ee,
                           const float* __restrict__ ss, float* __restrict__ outp,
                           long long total, int dshift, int dmask, int hshift, int hs, int M) {
  long long i = (long long)blockIdx.x * blockDim.x + threadIdx.x;
  if (i >= total) return;
  int e  = (int)(i >> dshift);
  int d4 = (int)i & dmask;
  int h  = d4 >> hshift;
  int s = src[e], d = dst[e];
  float alpha = ee[(e << hs) + h] / ss[(d << hs) + h];
  const float4 f = *(const float4*)(feat + (size_t)s * M + d4 * 4);
  float* op = outp + (size_t)d * M + d4 * 4;
  atomicAdd(op + 0, f.x * alpha);
  atomicAdd(op + 1, f.y * alpha);
  atomicAdd(op + 2, f.z * alpha);
  atomicAdd(op + 3, f.w * alpha);
}

static inline int ilog2i(int x) { int r = 0; while ((1 << r) < x) ++r; return r; }

// ---------------------------------------------------------------------------
extern "C" void kernel_launch(void* const* d_in, const int* in_sizes, int n_in,
                              void* d_out, int out_size, void* d_ws, size_t ws_size,
                              hipStream_t stream) {
  (void)in_sizes; (void)n_in; (void)out_size; (void)ws_size;
  const float* inputs = (const float*)d_in[0];
  const float* W0  = (const float*)d_in[1];
  const float* al0 = (const float*)d_in[2];
  const float* ar0 = (const float*)d_in[3];
  const float* b0  = (const float*)d_in[4];
  const float* W1  = (const float*)d_in[5];
  const float* al1 = (const float*)d_in[6];
  const float* ar1 = (const float*)d_in[7];
  const float* b1  = (const float*)d_in[8];
  const float* W2  = (const float*)d_in[9];
  const float* al2 = (const float*)d_in[10];
  const float* ar2 = (const float*)d_in[11];
  const float* b2  = (const float*)d_in[12];
  const int*   src = (const int*)d_in[13];
  const int*   dst = (const int*)d_in[14];
  float* out = (float*)d_out;

  char* ws = (char*)d_ws;
  size_t off = 0;
  auto alloc = [&](size_t bytes) -> void* {
    void* p = ws + off;
    off = (off + bytes + 255) & ~(size_t)255;
    return p;
  };
  float*    feat = (float*)alloc((size_t)NN * 256 * sizeof(float));
  float*    bufA = (float*)alloc((size_t)NN * 256 * sizeof(float));
  _Float16* h16  = (_Float16*)alloc((size_t)NN * 256 * sizeof(_Float16));
  _Float16* wt   = (_Float16*)alloc((size_t)256 * 256 * sizeof(_Float16));
  float*    el   = (float*)alloc((size_t)NN * 4 * sizeof(float));
  float*    er   = (float*)alloc((size_t)NN * 4 * sizeof(float));
  float*    mm   = (float*)alloc((size_t)NN * 4 * sizeof(float));
  float*    ssum = (float*)alloc((size_t)NN * 4 * sizeof(float));
  float*    ee   = (float*)alloc((size_t)NE * 4 * sizeof(float));

  auto run_layer = [&](const float* h, int K, const float* W, const float* al,
                       const float* ar, const float* b, int H, int D, float* outp) {
    const int M = H * D;
    // h (f32) -> h16; after this, h's buffer is dead and may alias outp
    { int n = NN * K;
      k_f32_to_f16<<<(n + 255) / 256, 256, 0, stream>>>(h, h16, n); }
    // W [K,M] -> Wt [M,K] f16
    { int n = K * M;
      k_transpose_f16<<<(n + 255) / 256, 256, 0, stream>>>(W, wt, K, M); }
    // feat = h @ W via WMMA
    if (M >= 128) {
      int colGroups = M / 128;                       // MT = 8
      int waves = (NN / 16) * colGroups;
      int blocks = (waves * 32 + 255) / 256;
      k_gemm_wmma<8><<<blocks, 256, 0, stream>>>(h16, wt, feat, NN, K, M);
    } else {
      int waves = NN / 16;                           // MT = 1
      int blocks = (waves * 32 + 255) / 256;
      k_gemm_wmma<1><<<blocks, 256, 0, stream>>>(h16, wt, feat, NN, K, M);
    }
    // attention logits per node
    { int n = NN * H;
      k_elr<<<(n + 255) / 256, 256, 0, stream>>>(feat, al, ar, el, er, NN, H, D); }
    // init segment max/sum and output (= bias)
    { int n = NN * H;
      k_init_ms<<<(n + 255) / 256, 256, 0, stream>>>(mm, ssum, n);
      int n2 = NN * M;
      k_init_out<<<(n2 + 255) / 256, 256, 0, stream>>>(b, outp, M - 1, n2); }
    const int hs = ilog2i(H);
    // edge softmax passes
    { int n = NE * H;
      k_edge_max<<<(n + 255) / 256, 256, 0, stream>>>(src, dst, el, er, mm, n, hs, H - 1);
      k_edge_exp<<<(n + 255) / 256, 256, 0, stream>>>(src, dst, el, er, mm, ssum, ee, n, hs, H - 1); }
    // weighted scatter-aggregate
    { long long n = (long long)NE * (M / 4);
      int dshift = ilog2i(M / 4);
      int hshift = ilog2i(D / 4);
      int blocks = (int)((n + 255) / 256);
      k_edge_agg<<<blocks, 256, 0, stream>>>(src, dst, feat, ee, ssum, outp,
                                             n, dshift, (M / 4) - 1, hshift, hs, M); }
  };

  // Layer 0: [N,128] -> [N,4,64];  Layer 1: [N,256] -> [N,4,64];  Layer 2: [N,256] -> [N,1,16]
  run_layer(inputs, 128, W0, al0, ar0, b0, 4, 64, bufA);
  run_layer(bufA,   256, W1, al1, ar1, b1, 4, 64, bufA);
  run_layer(bufA,   256, W2, al2, ar2, b2, 1, 16, out);   // H=1 -> mean over heads is identity
}
